// SparseMultiHeadAttention_14920716386408
// MI455X (gfx1250) — matrix operand order repaired and checked by reading, not compiled, not run on hardware
//
#include <hip/hip_runtime.h>
#include <hip/hip_bf16.h>
#include <math.h>

typedef __attribute__((ext_vector_type(16))) _Float16 v16h;
typedef __attribute__((ext_vector_type(8)))  _Float16 h8;
typedef __attribute__((ext_vector_type(8)))  float    v8f;
typedef __attribute__((ext_vector_type(4)))  unsigned int u32x4;
typedef __attribute__((ext_vector_type(8)))  int i32x8;
typedef __attribute__((ext_vector_type(4)))  int i32x4;

#define NTOK   4096
#define CCH    1024
#define NH     16
#define HD     64
#define QKV_LD 3072

union H16 { v16h v; h8 h[2]; };

__device__ __forceinline__ v8f wmma16(v16h a, v16h b, v8f c) {
  // D = A(16x32 f16) * B(32x16 f16) + C(16x16 f32)
  return __builtin_amdgcn_wmma_f32_16x16x32_f16(false, a, false, b, (short)0, c,
                                                false, false);
}

__device__ __forceinline__ h8 cvt8(const float* p) {
  h8 r;
#pragma unroll
  for (int i = 0; i < 8; ++i) r[i] = (_Float16)p[i];
  return r;
}

// 16 contiguous halves (two 16B LDS reads)
__device__ __forceinline__ v16h ld16(const _Float16* p) {
  H16 u;
  u.h[0] = *(const h8*)p;
  u.h[1] = *(const h8*)(p + 8);
  return u.v;
}

// A-matrix fragment (16x32 f16) from row-major LDS tile (ISA 7.12.2 layout).
__device__ __forceinline__ v16h ldAfrag(const _Float16* base, int stride,
                                        int row0, int lane) {
  const _Float16* r = base + (size_t)(row0 + (lane & 15)) * stride +
                      ((lane < 16) ? 0 : 8);
  H16 u;
  u.h[0] = *(const h8*)r;
  u.h[1] = *(const h8*)(r + 16);
  return u.v;
}

// token id for (window w, in-window index l) on the regular 16^3 grid
__device__ __forceinline__ int tok_of(int w, int l) {
  int wz = w >> 2, wy = (w >> 1) & 1, wx = w & 1;
  int lz = l >> 6, ly = (l >> 3) & 7, lx = l & 7;
  int z = wz * 8 + lz, y = wy * 8 + ly, x = wx * 8 + lx;
  return (z * 16 + y) * 16 + x;
}

// ---------------------------------------------------------------------------
// Tensor Data Mover: 2D f16 tile (tile1 rows x tile0 contiguous elems) from
// global to LDS, with optional LDS row padding.  D# layout per CDNA5 ISA §8.
// ---------------------------------------------------------------------------
#if __has_builtin(__builtin_amdgcn_tensor_load_to_lds)
#define HAVE_TDM 1
#else
#define HAVE_TDM 0
#endif

#if HAVE_TDM
__device__ __forceinline__ void tdm_load_2d_f16(unsigned lds_addr,
                                                const _Float16* gptr,
                                                unsigned tile0, unsigned tile1,
                                                unsigned long stride0,
                                                int pad, unsigned pad_interval,
                                                unsigned pad_amount) {
  unsigned long ga = (unsigned long)(const void*)gptr;
  u32x4 g0 = {0u, 0u, 0u, 0u};
  g0[0] = 1u;                                   // count=1, user descriptor
  g0[1] = lds_addr;                             // LDS byte address
  g0[2] = (unsigned)(ga & 0xffffffffu);         // global_addr[31:0]
  g0[3] = (unsigned)((ga >> 32) & 0x01ffffffu)  // global_addr[56:32]
          | (2u << 30);                         // type = 2 ("image")
  unsigned dw0 = (1u << 16);                    // data_size = 1 -> 2 bytes
  if (pad) dw0 |= (1u << 20)                    // pad_enable
               | (pad_interval << 22)           // 2^(code+1) DWORDs
               | (pad_amount << 25);            // (code+1) DWORDs
  unsigned td0 = tile0, td1 = tile1;            // tensor dims == tile dims
  i32x8 g1 = {0, 0, 0, 0, 0, 0, 0, 0};
  g1[0] = (int)dw0;
  g1[1] = (int)((td0 & 0xffffu) << 16);                    // tensor_dim0 lo
  g1[2] = (int)(((td0 >> 16) & 0xffffu) | ((td1 & 0xffffu) << 16));
  g1[3] = (int)(((td1 >> 16) & 0xffffu) | ((tile0 & 0xffffu) << 16)); // tile0
  g1[4] = (int)(tile1 & 0xffffu);                          // tile1, tile2=0
  g1[5] = (int)(unsigned)(stride0 & 0xffffffffu);          // dim0 stride lo
  g1[6] = (int)(unsigned)((stride0 >> 32) & 0xffffu);      // dim0 stride hi
  i32x4 z4 = {0, 0, 0, 0};
#if defined(__clang_major__) && (__clang_major__ >= 23)
  i32x8 z8 = {0, 0, 0, 0, 0, 0, 0, 0};
  __builtin_amdgcn_tensor_load_to_lds(g0, g1, z4, z4, z8, 0);
#else
  __builtin_amdgcn_tensor_load_to_lds(g0, g1, z4, z4, 0);
#endif
}
#endif

// ---------------------------------------------------------------------------
// f32 -> f16 conversion (operand pre-conversion; done once per matrix)
// ---------------------------------------------------------------------------
__global__ __launch_bounds__(256) void cvt_f32_f16(const float* __restrict__ src,
                                                   _Float16* __restrict__ dst,
                                                   int n4) {
  int i = blockIdx.x * blockDim.x + threadIdx.x;
  if (i >= n4) return;
  const float4 f = ((const float4*)src)[i];
  _Float16* d = dst + 4 * (size_t)i;
  d[0] = (_Float16)f.x; d[1] = (_Float16)f.y;
  d[2] = (_Float16)f.z; d[3] = (_Float16)f.w;
}

// ---------------------------------------------------------------------------
// C[M,N] = A[M,K] @ B[K,N] + bias ; A,B are f16, C f32.
// M,N multiples of 128, K multiple of 32.
// Double-buffered LDS tiles filled by the Tensor Data Mover: the next K-tile's
// tensor_load_to_lds is issued before computing the current tile, so the DMA
// overlaps the WMMA work; TENSORcnt is drained only at the next iteration top.
// ---------------------------------------------------------------------------
__global__ __launch_bounds__(256) void gemm_f16_wmma(
    const _Float16* __restrict__ A, const _Float16* __restrict__ B,
    const float* __restrict__ bias, float* __restrict__ C,
    int M, int N, int K) {
  __shared__ _Float16 At[2][128 * 40];   // 128 rows x 32 K  (pad 8 halves)
  __shared__ _Float16 Bt[2][32 * 136];   // 32 K x 128 cols  (pad 8 halves)
  const int tid  = threadIdx.x;
  const int lane = tid & 31;
  const int wave = tid >> 5;
  const int wm   = wave >> 1;   // 0..3 -> 32-row slab
  const int wn   = wave & 1;    // 0..1 -> 64-col slab
  const int mBase = blockIdx.y * 128;
  const int nBase = blockIdx.x * 128;

  const v8f vzero = {};
  v8f acc[2][4];
#pragma unroll
  for (int mi = 0; mi < 2; ++mi)
#pragma unroll
    for (int ni = 0; ni < 4; ++ni) acc[mi][ni] = vzero;

  const int nk = K >> 5;   // K/32 tiles

#if HAVE_TDM
  // prologue: fill buffer 0
  if (wave == 0) {
    tdm_load_2d_f16((unsigned)(size_t)(void*)&At[0][0],
                    A + (size_t)mBase * K, 32u, 128u,
                    (unsigned long)K, 1, 3u, 3u);   // LDS stride 40 halves
    tdm_load_2d_f16((unsigned)(size_t)(void*)&Bt[0][0],
                    B + nBase, 128u, 32u,
                    (unsigned long)N, 1, 5u, 3u);   // LDS stride 136 halves
  }
#endif

  for (int it = 0; it < nk; ++it) {
    const int k0 = it << 5;
#if HAVE_TDM
    if (wave == 0) __builtin_amdgcn_s_wait_tensorcnt(0);  // buf[it&1] filled
    __syncthreads();   // all waves: buf[it&1] ready AND buf[(it+1)&1] free
    if (wave == 0 && (it + 1) < nk) {
      // overlap: DMA next tile while this tile is being consumed
      tdm_load_2d_f16((unsigned)(size_t)(void*)&At[(it + 1) & 1][0],
                      A + (size_t)mBase * K + (k0 + 32), 32u, 128u,
                      (unsigned long)K, 1, 3u, 3u);
      tdm_load_2d_f16((unsigned)(size_t)(void*)&Bt[(it + 1) & 1][0],
                      B + (size_t)(k0 + 32) * N + nBase, 128u, 32u,
                      (unsigned long)N, 1, 5u, 3u);
    }
#else
    __syncthreads();
    for (int i = tid; i < 512; i += 256) {          // A: 128x32 halves
      int r = i >> 2, c = (i & 3) << 3;
      *(h8*)&At[0][r * 40 + c] =
          *(const h8*)(A + (size_t)(mBase + r) * K + k0 + c);
    }
    for (int i = tid; i < 512; i += 256) {          // B: 32x128 halves
      int r = i >> 4, c = (i & 15) << 3;
      *(h8*)&Bt[0][r * 136 + c] =
          *(const h8*)(B + (size_t)(k0 + r) * N + nBase + c);
    }
    __syncthreads();
#endif
    if (k0 + 64 < K) {  // warm the tile after next in L2 (global_prefetch_b8)
      __builtin_prefetch((const void*)(A + (size_t)(mBase + tid) * K + k0 + 64));
      __builtin_prefetch((const void*)(B + (size_t)(k0 + 64 + (tid >> 3)) * N + nBase));
    }

#if HAVE_TDM
    const _Float16* Ab = &At[it & 1][0];
    const _Float16* Bb = &Bt[it & 1][0];
#else
    const _Float16* Ab = &At[0][0];
    const _Float16* Bb = &Bt[0][0];
#endif
    v16h aF[2];
#pragma unroll
    for (int mi = 0; mi < 2; ++mi)
      aF[mi] = ldAfrag(Ab, 40, wm * 32 + mi * 16, lane);
#pragma unroll
    for (int ni = 0; ni < 4; ++ni) {
      // B fragment: lane holds N-col lane&15, K in A-fragment (table) order
      v16h bF;
#pragma unroll
      for (int s = 0; s < 16; ++s)
        bF[s] = Bb[(((lane < 16) ? 0 : 8) + s + (s & 8)) * 136 + wn * 64 + ni * 16 + (lane & 15)];
#pragma unroll
      for (int mi = 0; mi < 2; ++mi)
        acc[mi][ni] = wmma16(aF[mi], bF, acc[mi][ni]);
    }
  }

  // C/D layout: vgpr i, lanes 0-15 -> (M=i, N=lane); lanes 16-31 -> (M=i+8)
#pragma unroll
  for (int mi = 0; mi < 2; ++mi) {
    int rbase = mBase + wm * 32 + mi * 16 + 8 * (lane >> 4);
#pragma unroll
    for (int ni = 0; ni < 4; ++ni) {
      int col = nBase + wn * 64 + ni * 16 + (lane & 15);
      float bb = bias[col];
#pragma unroll
      for (int i = 0; i < 8; ++i)
        C[(size_t)(rbase + i) * N + col] = acc[mi][ni][i] + bb;
    }
  }
}

// ---------------------------------------------------------------------------
// Fused multi-head RMS-norm (q,k) + 3D RoPE, in place on qkv buffer (f32).
// One thread per (token, head).
// ---------------------------------------------------------------------------
__global__ __launch_bounds__(256) void normrope_kernel(
    float* __restrict__ qkv, const int* __restrict__ coords,
    const float* __restrict__ qg, const float* __restrict__ kg) {
  int t = blockIdx.x * blockDim.x + threadIdx.x;
  if (t >= NTOK * NH) return;
  int n = t >> 4, h = t & 15;
  float c3[3];
  c3[0] = (float)coords[n * 4 + 1];
  c3[1] = (float)coords[n * 4 + 2];
  c3[2] = (float)coords[n * 4 + 3];
  float cs[32], sn[32];
#pragma unroll
  for (int p = 0; p < 32; ++p) {
    float ang = 0.0f;
    if (p < 30) {
      int a = p / 10, f = p % 10;
      // freq = 10000^(-f/10)
      float freq = __expf(-(float)f * 0.1f * 9.210340371976184f);
      ang = c3[a] * freq;
    }
    cs[p] = __cosf(ang);
    sn[p] = __sinf(ang);
  }
#pragma unroll
  for (int s = 0; s < 2; ++s) {
    float* row = qkv + (size_t)n * QKV_LD + s * CCH + h * HD;
    const float* g = (s == 0 ? qg : kg) + h * HD;
    float v[64];
    float ss = 0.0f;
#pragma unroll
    for (int d = 0; d < 64; ++d) { v[d] = row[d]; ss += v[d] * v[d]; }
    float scale = 8.0f / fmaxf(sqrtf(ss), 1e-12f);  // sqrt(D)=8
#pragma unroll
    for (int d = 0; d < 64; ++d) v[d] *= scale * g[d];
#pragma unroll
    for (int p = 0; p < 32; ++p) {
      float re = v[2 * p], im = v[2 * p + 1];
      row[2 * p]     = re * cs[p] - im * sn[p];
      row[2 * p + 1] = re * sn[p] + im * cs[p];
    }
  }
}

// ---------------------------------------------------------------------------
// Flash-style windowed attention. Grid (qchunk=4, head=16, window=8).
// Each of 8 waves owns a 16-row Q tile; K/V staged 64 keys at a time in LDS.
// Output written as f16 (A-operand of the out-projection GEMM).
// ---------------------------------------------------------------------------
__global__ __launch_bounds__(256) void attn_wmma(const float* __restrict__ qkv,
                                                 _Float16* __restrict__ obuf) {
  __shared__ _Float16 KtS[64][72];       // K: [key][d], f16
  __shared__ _Float16 VS[64][72];        // V^T: [d][key], f16
  __shared__ _Float16 Pscr[8][16][32];   // per-wave P scratch (C->A relayout)

  const int tid  = threadIdx.x;
  const int lane = tid & 31;
  const int wv   = tid >> 5;
  const int qc = blockIdx.x;   // 0..3 : 128-row q chunk
  const int h  = blockIdx.y;   // 0..15
  const int w  = blockIdx.z;   // 0..7

  // --- Q A-fragments (2 x K=32 over d), straight from global f32 -----------
  const int mrow = lane & 15;
  const int c0   = (lane < 16) ? 0 : 8;
  const int ql   = qc * 128 + wv * 16 + mrow;
  const int qn   = tok_of(w, ql);
  const float* qrow = qkv + (size_t)qn * QKV_LD + h * HD;
  v16h Qa[2];
#pragma unroll
  for (int kc = 0; kc < 2; ++kc) {
    H16 u;
    u.h[0] = cvt8(qrow + kc * 32 + c0);
    u.h[1] = cvt8(qrow + kc * 32 + c0 + 16);
    Qa[kc] = u.v;
  }

  const v8f vzero = {};
  v8f O[4];                       // 16 x 64 f32 accumulator (4 n-tiles)
  float mrun[8], lrun[8];
#pragma unroll
  for (int t = 0; t < 4; ++t) O[t] = vzero;
#pragma unroll
  for (int i = 0; i < 8; ++i) { mrun[i] = -INFINITY; lrun[i] = 0.0f; }

  for (int js = 0; js < 8; ++js) {        // 8 stages x 64 keys
    __syncthreads();
    for (int i = tid; i < 1024; i += 256) {
      int key = i >> 4;
      int dd  = (i & 15) << 2;
      int kn  = tok_of(w, js * 64 + key);
      const float4 kf = *(const float4*)(qkv + (size_t)kn * QKV_LD + CCH + h * HD + dd);
      KtS[key][dd + 0] = (_Float16)kf.x;
      KtS[key][dd + 1] = (_Float16)kf.y;
      KtS[key][dd + 2] = (_Float16)kf.z;
      KtS[key][dd + 3] = (_Float16)kf.w;
      const float4 vf = *(const float4*)(qkv + (size_t)kn * QKV_LD + 2 * CCH + h * HD + dd);
      VS[dd + 0][key] = (_Float16)vf.x;
      VS[dd + 1][key] = (_Float16)vf.y;
      VS[dd + 2][key] = (_Float16)vf.z;
      VS[dd + 3][key] = (_Float16)vf.w;
    }
    __syncthreads();

#pragma unroll
    for (int jj = 0; jj < 2; ++jj) {      // 32-key sub-blocks
      const int key0 = jj * 32;
      // S = Q @ K^T  (two 16x16 tiles, each K=64 via two WMMAs)
      v8f s0 = vzero, s1 = vzero;
      s0 = wmma16(Qa[0], ldAfrag(&KtS[0][0], 72, key0, lane),       s0);
      s0 = wmma16(Qa[1], ldAfrag(&KtS[0][32], 72, key0, lane),      s0);
      s1 = wmma16(Qa[0], ldAfrag(&KtS[0][0], 72, key0 + 16, lane),  s1);
      s1 = wmma16(Qa[1], ldAfrag(&KtS[0][32], 72, key0 + 16, lane), s1);

      float al[8];
#pragma unroll
      for (int i = 0; i < 8; ++i) {
        float x0 = s0[i] * 0.125f;        // 1/sqrt(64)
        float x1 = s1[i] * 0.125f;
        float mx = fmaxf(x0, x1);
#pragma unroll
        for (int msk = 8; msk >= 1; msk >>= 1)
          mx = fmaxf(mx, __shfl_xor(mx, msk, 32));  // row-wise over 16 lanes
        float mnew = fmaxf(mrun[i], mx);
        float a  = __expf(mrun[i] - mnew);
        float p0 = __expf(x0 - mnew);
        float p1 = __expf(x1 - mnew);
        float rs = p0 + p1;
#pragma unroll
        for (int msk = 8; msk >= 1; msk >>= 1)
          rs += __shfl_xor(rs, msk, 32);
        lrun[i] = lrun[i] * a + rs;
        mrun[i] = mnew;
        al[i]   = a;
        int prow = i + 8 * (lane >> 4);
        Pscr[wv][prow][lane & 15]        = (_Float16)p0;
        Pscr[wv][prow][(lane & 15) + 16] = (_Float16)p1;
      }
#pragma unroll
      for (int t = 0; t < 4; ++t)
#pragma unroll
        for (int i = 0; i < 8; ++i) O[t][i] = O[t][i] * al[i];

      __syncthreads();   // order per-wave P scratch (uniform barrier)

      // P as A-fragment (K = 32 keys)
      H16 up;
      up.h[0] = *(const h8*)&Pscr[wv][mrow][c0];
      up.h[1] = *(const h8*)&Pscr[wv][mrow][c0 + 16];
      v16h Pa = up.v;
      // O += P @ V  (B fragment: lane holds d-col lane&15, keys in P's A order)
#pragma unroll
      for (int t = 0; t < 4; ++t)
        O[t] = wmma16(Pa, ldAfrag(&VS[0][key0], 72, t * 16, lane), O[t]);
    }
  }

  // normalize and scatter back to token order (o[inv] is analytic); f16 out
  float inv[8];
#pragma unroll
  for (int i = 0; i < 8; ++i) inv[i] = 1.0f / lrun[i];
#pragma unroll
  for (int t = 0; t < 4; ++t)
#pragma unroll
    for (int i = 0; i < 8; ++i) {
      int l = qc * 128 + wv * 16 + i + 8 * (lane >> 4);
      int n = tok_of(w, l);
      int col = h * HD + t * 16 + (lane & 15);
      obuf[(size_t)n * CCH + col] = (_Float16)(O[t][i] * inv[i]);
    }
}

// ---------------------------------------------------------------------------
extern "C" void kernel_launch(void* const* d_in, const int* in_sizes, int n_in,
                              void* d_out, int out_size, void* d_ws, size_t ws_size,
                              hipStream_t stream) {
  (void)in_sizes; (void)n_in; (void)out_size; (void)ws_size;
  const float* x      = (const float*)d_in[0];
  const int*   coords = (const int*)d_in[1];
  const float* w_qkv  = (const float*)d_in[2];
  const float* b_qkv  = (const float*)d_in[3];
  const float* q_g    = (const float*)d_in[4];
  const float* k_g    = (const float*)d_in[5];
  const float* w_out  = (const float*)d_in[6];
  const float* b_out  = (const float*)d_in[7];
  float* out = (float*)d_out;

  char* ws = (char*)d_ws;
  float*    qkvbuf = (float*)(ws + 0);                    // 4096x3072 f32 (48MB)
  _Float16* xh     = (_Float16*)(ws + 50331648);          // 4096x1024 f16
  _Float16* wqkvh  = (_Float16*)(ws + 58720256);          // 1024x3072 f16
  _Float16* wouth  = (_Float16*)(ws + 65011712);          // 1024x1024 f16
  _Float16* obufh  = (_Float16*)(ws + 67108864);          // 4096x1024 f16

  // one-time operand pre-conversion to f16
  cvt_f32_f16<<<(NTOK * CCH / 4) / 256, 256, 0, stream>>>(x, xh, NTOK * CCH / 4);
  cvt_f32_f16<<<(CCH * QKV_LD / 4) / 256, 256, 0, stream>>>(w_qkv, wqkvh,
                                                            CCH * QKV_LD / 4);
  cvt_f32_f16<<<(CCH * CCH / 4) / 256, 256, 0, stream>>>(w_out, wouth,
                                                         CCH * CCH / 4);
  // qkv = x @ w_qkv + b_qkv
  gemm_f16_wmma<<<dim3(QKV_LD / 128, NTOK / 128), 256, 0, stream>>>(
      xh, wqkvh, b_qkv, qkvbuf, NTOK, QKV_LD, CCH);
  // RMS-norm + RoPE on q,k (in place)
  normrope_kernel<<<(NTOK * NH) / 256, 256, 0, stream>>>(qkvbuf, coords, q_g, k_g);
  // windowed flash attention -> f16 activations in token order
  attn_wmma<<<dim3(4, NH, 8), 256, 0, stream>>>(qkvbuf, obufh);
  // out = o @ w_out + b_out
  gemm_f16_wmma<<<dim3(CCH / 128, NTOK / 128), 256, 0, stream>>>(
      obufh, wouth, b_out, out, NTOK, CCH, CCH);
}
